// EmbeddingRNN_44590350467221
// MI455X (gfx1250) — compile-verified, matrix operand order checked by reading, not run on hardware
//
#include <hip/hip_runtime.h>
#include <hip/hip_bf16.h>

// CDNA5 / gfx1250 fused embedding + 16-step LSTM scan using bf16 WMMA.
//
// x (64,128,16) int, emb (155,256) f32, W (256,1024) f32, U (256,1024) f32,
// b (1024) f32  ->  out (64,128,256) f32.
//
// Row-independent recurrence: each block privately scans 32 rows over 16
// timesteps. c-state in VGPRs, h-state in LDS (bf16), W/U prepacked into
// WMMA B-fragment order in workspace (bf16) and streamed from L2 each step.
// The rolled kc loop + an opaque (asm-laundered) integer offset keep the
// t-invariant B fragments from being hoisted/spilled, while preserving the
// global address space so the loads lower to global_load_b128 (saddr form),
// not flat_load (round-2 issue: flat ties up DScnt and the LDS path).

typedef __attribute__((ext_vector_type(16))) __bf16 v16bf;
typedef __attribute__((ext_vector_type(8)))  __bf16 v8bf;
typedef __attribute__((ext_vector_type(8)))  float  v8f;

#define DD   256          // hidden size
#define HSTR 264          // LDS h row stride in bf16 elements (256 + 8 pad)

// K index inside a 16x16x32 bf16 fragment for (lane, element e):
// lanes 0-15 hold K 0-7 (VGPR0-3) and K 16-23 (VGPR4-7); lanes 16-31 are +8.
__device__ __forceinline__ int kmap_e(int lane, int e) {
  int hi = (lane & 16) ? 8 : 0;
  return (e < 8) ? (e + hi) : (16 + (e - 8) + hi);
}

__global__ void cvt_bf16_kernel(const float* __restrict__ src,
                                __bf16* __restrict__ dst, int n) {
  int i = blockIdx.x * blockDim.x + threadIdx.x;
  if (i < n) dst[i] = (__bf16)src[i];
}

// Pack a (256 x 1024) row-major f32 matrix into WMMA-B fragment order, bf16.
// Fragment (kc, nt) occupies 32 lanes x 16 elements contiguously: one 32-byte
// load per lane in the main kernel.
__global__ void pack_b_kernel(const float* __restrict__ src,
                              __bf16* __restrict__ dst) {
  int tid  = blockIdx.x * blockDim.x + threadIdx.x;  // 0 .. 262143
  int e    = tid & 15;
  int lane = (tid >> 4) & 31;
  int nt   = (tid >> 9) & 63;
  int kc   = tid >> 15;
  int k = kc * 32 + kmap_e(lane, e);
  int n = nt * 16 + (lane & 15);
  dst[tid] = (__bf16)src[k * 1024 + n];
}

// Build an A fragment from two contiguous 8-element (16B) halves.
__device__ __forceinline__ v16bf load_frag(const __bf16* lo, const __bf16* hi) {
  v8bf a = *(const v8bf*)lo;
  v8bf b = *(const v8bf*)hi;
  v16bf r;
#pragma unroll
  for (int i = 0; i < 8; ++i) { r[i] = a[i]; r[i + 8] = b[i]; }
  return r;
}

__device__ __forceinline__ float sigmoidf_(float z) {
  return 1.0f / (1.0f + __expf(-z));
}

__global__ __launch_bounds__(512) void lstm_fused_kernel(
    const int* __restrict__ x, const __bf16* __restrict__ embB,
    const __bf16* __restrict__ Wp, const __bf16* __restrict__ Up,
    const float* __restrict__ bvec, float* __restrict__ out) {
  __shared__ __bf16 hlds[32 * HSTR];

  const int tid  = threadIdx.x;
  const int lane = tid & 31;
  const int w    = tid >> 5;          // wave 0..15
  const int m    = lane & 15;         // row within a 16-row subtile
  const int hiK  = (lane & 16) ? 8 : 0;
  const int blockRow = blockIdx.x;    // 0..255, 32 rows each

  // zero initial h
  for (int i = tid; i < 32 * HSTR; i += 512) hlds[i] = (__bf16)0.0f;

  // wave w owns N columns [w*16, w*16+16) within each of the 4 gate blocks
  int ntb[4]; float bias[4];
#pragma unroll
  for (int g = 0; g < 4; ++g) {
    ntb[g]  = g * 16 + w;                       // N-tile index into packed B
    bias[g] = bvec[g * 256 + w * 16 + m];
  }

  v8f cst[2];                         // cell state: 2 M-subtiles x 16 cols
#pragma unroll
  for (int mt = 0; mt < 2; ++mt)
#pragma unroll
    for (int r = 0; r < 8; ++r) cst[mt][r] = 0.0f;

  const int* xr0 = x + (blockRow * 32 + m) * 16;        // char idx, M-tile 0
  const int* xr1 = x + (blockRow * 32 + 16 + m) * 16;   // char idx, M-tile 1
  const __bf16* hr0 = hlds + m * HSTR;
  const __bf16* hr1 = hlds + (16 + m) * HSTR;

  __syncthreads();

#pragma unroll 1
  for (int t = 0; t < 16; ++t) {
    const __bf16* e0 = embB + xr0[t] * DD;
    const __bf16* e1 = embB + xr1[t] * DD;

    // Opaque zero offset: blocks cross-iteration CSE/LICM of the weight
    // fragment loads while keeping pointer provenance (=> global_load saddr).
    size_t zoff = 0;
    asm volatile("" : "+s"(zoff));
    const __bf16* WpT = Wp + zoff;
    const __bf16* UpT = Up + zoff;

    v8f acc[8];                       // [gate*2 + mt]
#pragma unroll
    for (int j = 0; j < 8; ++j)
#pragma unroll
      for (int r = 0; r < 8; ++r) acc[j][r] = 0.0f;

#pragma unroll 1
    for (int kc = 0; kc < 8; ++kc) {  // K = 256 in chunks of 32
      const int k0 = kc * 32 + hiK;

      // W phase: A = gathered embedding rows
      v16bf ae0 = load_frag(e0 + k0, e0 + k0 + 16);
      v16bf ae1 = load_frag(e1 + k0, e1 + k0 + 16);
#pragma unroll
      for (int g = 0; g < 4; ++g) {
        v16bf bw = *(const v16bf*)(WpT + ((kc * 64 + ntb[g]) * 32 + lane) * 16);
        acc[g * 2 + 0] = __builtin_amdgcn_wmma_f32_16x16x32_bf16(
            false, ae0, false, bw, (short)0, acc[g * 2 + 0], false, false);
        acc[g * 2 + 1] = __builtin_amdgcn_wmma_f32_16x16x32_bf16(
            false, ae1, false, bw, (short)0, acc[g * 2 + 1], false, false);
      }

      // U phase: A = h(t-1) from LDS
      v16bf ah0 = load_frag(hr0 + k0, hr0 + k0 + 16);
      v16bf ah1 = load_frag(hr1 + k0, hr1 + k0 + 16);
#pragma unroll
      for (int g = 0; g < 4; ++g) {
        v16bf bu = *(const v16bf*)(UpT + ((kc * 64 + ntb[g]) * 32 + lane) * 16);
        acc[g * 2 + 0] = __builtin_amdgcn_wmma_f32_16x16x32_bf16(
            false, ah0, false, bu, (short)0, acc[g * 2 + 0], false, false);
        acc[g * 2 + 1] = __builtin_amdgcn_wmma_f32_16x16x32_bf16(
            false, ah1, false, bu, (short)0, acc[g * 2 + 1], false, false);
      }
    }

    __syncthreads();  // all waves finished reading h(t-1)

    const int hcol = w * 16 + m;      // this lane's h column
#pragma unroll
    for (int mt = 0; mt < 2; ++mt) {
      v8f zi = acc[0 + mt] + bias[0];
      v8f zf = acc[2 + mt] + bias[1];
      v8f zg = acc[4 + mt] + bias[2];
      v8f zo = acc[6 + mt] + bias[3];
#pragma unroll
      for (int r = 0; r < 8; ++r) {
        float ig = sigmoidf_(zi[r]);
        float fg = sigmoidf_(zf[r]);
        float og = sigmoidf_(zo[r]);
        float cc = fg * cst[mt][r] + ig * zg[r];    // identity candidate
        float hh = og * cc;                         // identity output act
        cst[mt][r] = cc;
        const int row = mt * 16 + r + hiK;          // C-tile row mapping
        hlds[row * HSTR + hcol] = (__bf16)hh;
        if (t == 15)
          out[(blockRow * 32 + row) * DD + hcol] = hh;
      }
    }
    __syncthreads();  // h(t) visible before next step's reads
  }
}

extern "C" void kernel_launch(void* const* d_in, const int* in_sizes, int n_in,
                              void* d_out, int out_size, void* d_ws, size_t ws_size,
                              hipStream_t stream) {
  const int*   x   = (const int*)d_in[0];    // (64,128,16)
  const float* emb = (const float*)d_in[1];  // (155,256)
  const float* W   = (const float*)d_in[2];  // (256,1024)
  const float* U   = (const float*)d_in[3];  // (256,1024)
  const float* b   = (const float*)d_in[4];  // (1024,)
  float* out = (float*)d_out;                // (64,128,256)

  // workspace layout (bf16): emb (padded to 40960) | Wp (262144) | Up (262144)
  __bf16* embB = (__bf16*)d_ws;
  __bf16* Wp   = embB + 40960;
  __bf16* Up   = Wp + 262144;

  cvt_bf16_kernel<<<(155 * 256 + 255) / 256, 256, 0, stream>>>(emb, embB, 155 * 256);
  pack_b_kernel<<<1024, 256, 0, stream>>>(W, Wp);
  pack_b_kernel<<<1024, 256, 0, stream>>>(U, Up);
  lstm_fused_kernel<<<256, 512, 0, stream>>>(x, embB, Wp, Up, b, out);
}
